// KANInterpoLayer_15968688407294
// MI455X (gfx1250) — matrix operand order repaired and checked by reading, not compiled, not run on hardware
//
#include <hip/hip_runtime.h>

typedef __attribute__((ext_vector_type(16))) _Float16 v16h;
typedef __attribute__((ext_vector_type(8)))  float    v8f;

#define B_SZ   1024
#define I_SZ   256
#define J_SZ   256
#define K_SZ   64

#define NSPLIT 16
#define ICHUNK (I_SZ / NSPLIT)      // 16 i per split
#define BTILE  64                   // rows of out per workgroup (4 x 16 subtiles)

#define FRAG_HALVES  512            // 32 lanes * 16 halves per 32x16 B fragment
#define FRAGS_PER_I  32             // 2 k-tiles * 16 j-tiles
#define I_STRIDE     ((size_t)FRAGS_PER_I * FRAG_HALVES)        // halves per i
#define YT_HALVES    ((size_t)I_SZ * I_STRIDE)                  // 4M halves = 8MB
#define YT_BYTES     (YT_HALVES * sizeof(_Float16))

// ---------------------------------------------------------------------------
// Kernel 1: Y [I,J,K] f32 -> pre-swizzled WMMA B fragments (f16).
// Fragment (i,kt,jt): lane l, half h holds Y[i, jt*16 + (l&15), kt*32 + (l<16?0:16) + h]
// Each thread handles one (i,kt,jt,lane): reads 64B contiguous, writes 32B contiguous.
// ---------------------------------------------------------------------------
__global__ __launch_bounds__(256) void kan_pack(const float* __restrict__ Y,
                                                _Float16* __restrict__ Yt) {
    const int tid = blockIdx.x * 256 + threadIdx.x;     // tid = ((i*2+kt)*16+jt)*32 + l
    const int l   = tid & 31;
    const int jt  = (tid >> 5) & 15;
    const int kt  = (tid >> 9) & 1;
    const int i   = tid >> 10;
    const int j   = jt * 16 + (l & 15);
    const int klo = (l & 16);                           // 0 or 16

    const float* src = Y + ((size_t)(i * J_SZ + j)) * K_SZ + kt * 32 + klo;

    float4 sv[4];
#pragma unroll
    for (int q = 0; q < 4; ++q) sv[q] = ((const float4*)src)[q];
    const float* sf = (const float*)sv;

    v16h d;
#pragma unroll
    for (int h = 0; h < 16; ++h) d[h] = (_Float16)sf[h];

    *(v16h*)(Yt + (size_t)tid * 16) = d;
}

// ---------------------------------------------------------------------------
// Kernel 2: split-K GEMM with on-the-fly densified interpolation weights.
// Block: 256 threads (8 wave32). BTILE=64: wave w -> b-subtile (w&3),
// j-tiles (w>>2)*8 .. +7.  Each A fragment feeds 8 WMMAs (16 per i per wave),
// so the cmp/cndmask densify co-executes behind the XDL pipe.
// ---------------------------------------------------------------------------
__global__ __launch_bounds__(256) void kan_gemm(const float* __restrict__ x,
                                                const float* __restrict__ Xg,
                                                const _Float16* __restrict__ Yt,
                                                float* __restrict__ partial) {
    const int lane  = threadIdx.x & 31;
    const int wave  = threadIdx.x >> 5;
    const int bsub  = wave & 3;
    const int jt0   = (wave >> 2) * 8;
    const int bbase = blockIdx.x * BTILE + bsub * 16;
    const int split = blockIdx.y;
    const int i0    = split * ICHUNK;

    const int mrow  = lane & 15;                 // b-row in subtile this lane gathers
    const int brow  = bbase + mrow;
    const int lane8 = (lane & 16) ? 8 : 0;       // A-layout lane k offset

    const float X0    = Xg[0];
    const float inv_h = 1.0f / (Xg[1] - Xg[0]);  // uniform linspace grid

    v8f acc[8] = {};

    const _Float16* frag = Yt + (size_t)i0 * I_STRIDE + (size_t)lane * 16;

    for (int i = i0; i < i0 + ICHUNK; ++i) {
        // --- interval search on uniform grid + lerp weight (exact x0/x1 from X) ---
        const float xv = x[(size_t)brow * I_SZ + i];
        int idx0 = (int)floorf((xv - X0) * inv_h);
        idx0 = idx0 < 0 ? 0 : (idx0 > (K_SZ - 2) ? (K_SZ - 2) : idx0);
        const float x0 = Xg[idx0];
        const float x1 = Xg[idx0 + 1];
        const float t  = (xv - x0) / (x1 - x0);
        const _Float16 w1 = (_Float16)t;
        const _Float16 w0 = (_Float16)(1.0f - t);

        // prefetch next i's fragment stream (this wave's j-tile region, both k-tiles)
        __builtin_prefetch(frag + I_STRIDE + (size_t)jt0 * FRAG_HALVES, 0, 1);
        __builtin_prefetch(frag + I_STRIDE + (size_t)(16 + jt0) * FRAG_HALVES, 0, 1);

#pragma unroll
        for (int kt = 0; kt < 2; ++kt) {
            const int r0 = idx0 - kt * 32;       // target k within this k-tile
            const int r1 = r0 + 1;

            // Densify sparse row of W into WMMA A fragment (16x32 f16 layout):
            // lane l, half h -> k = 2*lane8 + h
            v16h a;
#pragma unroll
            for (int h = 0; h < 16; ++h) {
                const int kA = (lane8 << 1) + h;
                a[h] = (kA == r0) ? w0 : ((kA == r1) ? w1 : (_Float16)0.0f);
            }

#pragma unroll
            for (int q = 0; q < 8; ++q) {
                const int jt = jt0 + q;
                const v16h bf = *(const v16h*)(frag + (size_t)(kt * 16 + jt) * FRAG_HALVES);
                acc[q] = __builtin_amdgcn_wmma_f32_16x16x32_f16(
                    false, a, false, bf, (short)0, acc[q], false, false);
            }
        }
        frag += I_STRIDE;
    }

    // Epilogue: C/D layout -> VGPR r: lanes 0-15 M=r, lanes 16-31 M=8+r; N = lane&15
    float* pout = partial + ((size_t)split * B_SZ + bbase) * J_SZ;
    const int mofs = (lane & 16) ? 8 : 0;
#pragma unroll
    for (int q = 0; q < 8; ++q) {
        const int jcol = (jt0 + q) * 16 + (lane & 15);
#pragma unroll
        for (int r = 0; r < 8; ++r) {
            pout[(size_t)(mofs + r) * J_SZ + jcol] = acc[q][r];
        }
    }
}

// ---------------------------------------------------------------------------
// Kernel 3: sum split-K partials (deterministic, no atomics).
// ---------------------------------------------------------------------------
__global__ __launch_bounds__(256) void kan_reduce(const float* __restrict__ partial,
                                                  float* __restrict__ out) {
    const int idx = blockIdx.x * 256 + threadIdx.x;
    float s = 0.0f;
#pragma unroll
    for (int p = 0; p < NSPLIT; ++p)
        s += partial[(size_t)p * (B_SZ * J_SZ) + idx];
    out[idx] = s;
}

extern "C" void kernel_launch(void* const* d_in, const int* in_sizes, int n_in,
                              void* d_out, int out_size, void* d_ws, size_t ws_size,
                              hipStream_t stream) {
    const float* x  = (const float*)d_in[0];   // [1024, 256]
    const float* Xg = (const float*)d_in[1];   // [64]
    const float* Y  = (const float*)d_in[2];   // [256, 256, 64]
    float* out      = (float*)d_out;           // [1024, 256]

    _Float16* Yt   = (_Float16*)d_ws;                        // 8 MB fragments
    float* partial = (float*)((char*)d_ws + YT_BYTES);       // 16 MB partials

    // 1) pack/convert Y into WMMA B fragments
    kan_pack<<<(I_SZ * FRAGS_PER_I * 32) / 256, 256, 0, stream>>>(Y, Yt);

    // 2) densified-W f16 WMMA GEMM with split-K over i
    dim3 grid(B_SZ / BTILE, NSPLIT);
    kan_gemm<<<grid, 256, 0, stream>>>(x, Xg, Yt, partial);

    // 3) reduce partials
    kan_reduce<<<(B_SZ * J_SZ) / 256, 256, 0, stream>>>(partial, out);
}
